// ScaledDotProductAttention_77592879169685
// MI455X (gfx1250) — matrix operand order repaired and checked by reading, not compiled, not run on hardware
//
#include <hip/hip_runtime.h>

typedef __attribute__((ext_vector_type(16))) _Float16 v16h;
typedef __attribute__((ext_vector_type(8)))  _Float16 v8h;
typedef __attribute__((ext_vector_type(8)))  float    v8f;
typedef __attribute__((ext_vector_type(4)))  float    v4f;
typedef __attribute__((ext_vector_type(2)))  float    v2f;

#define B_   16
#define L_   2048
#define D_   64
#define QT_PER_BATCH (L_/16)          // 128 query tiles per batch
#define N_ELEM (B_*L_*D_)             // 2097152
#define EPS_ 1e-7f

#if __has_builtin(__builtin_amdgcn_exp2f)
#define EXP2F(x) __builtin_amdgcn_exp2f(x)
#else
#define EXP2F(x) exp2f(x)
#endif

// Q prescale: 1/sqrt(64) * log2(e)  -> softmax computed in base 2, exactly
// equivalent: 2^(c*s) / sum 2^(c*s) == e^(s/8) / sum e^(s/8) for c = log2e/8.
#define QSCALE_ (0.125f * 1.4426950408889634f)

// ---------------- prep: fp32 -> (hi,lo) f16 split ----------------------------------
__global__ __launch_bounds__(256) void prep_split_f16(
    const float* __restrict__ q, const float* __restrict__ k,
    _Float16* __restrict__ qh, _Float16* __restrict__ ql,
    _Float16* __restrict__ kh, _Float16* __restrict__ kl) {
  int i = blockIdx.x * 256 + threadIdx.x;
  if (i >= N_ELEM) return;
  float qs = q[i] * QSCALE_;
  _Float16 qhi = (_Float16)qs;
  qh[i] = qhi;
  ql[i] = (_Float16)(qs - (float)qhi);
  float kv = k[i];
  _Float16 khi = (_Float16)kv;
  kh[i] = khi;
  kl[i] = (_Float16)(kv - (float)khi);
}

// branchless top-3 insert (v_cmp + v_cndmask only)
__device__ __forceinline__ void top3_insert(float s, int idx,
    float& t0, float& t1, float& t2, int& i0, int& i1, int& i2) {
  bool g0 = s > t0, g1 = s > t1, g2 = s > t2;
  float n1 = g0 ? t0 : s;      // value that lands in slot1 when g1
  int   m1 = g0 ? i0 : idx;
  t2 = g1 ? t1 : (g2 ? s : t2);
  i2 = g1 ? i1 : (g2 ? idx : i2);
  t1 = g1 ? n1 : t1;
  i1 = g1 ? m1 : i1;
  t0 = g0 ? s : t0;
  i0 = g0 ? idx : i0;
}

union AF { v16h v; v8h h[2]; };

// ---------------- main: one wave per 16-query tile ----------------------------------
__global__ __launch_bounds__(256) void sparse_attn_kernel(
    const _Float16* __restrict__ qh, const _Float16* __restrict__ ql,
    const _Float16* __restrict__ kh, const _Float16* __restrict__ kl,
    const float* __restrict__ v, float* __restrict__ out,
    float* __restrict__ attnw) {
  __shared__ __align__(16) float sh[8][16][20];   // per-wave 16x16 tile, stride 20
  const int wave = threadIdx.x >> 5;
  const int lane = threadIdx.x & 31;
  const int l16  = lane & 15;
  const int hs   = lane >> 4;                     // half-wave select

  const int tile  = blockIdx.x * 8 + wave;        // 2048 tiles total
  const int b     = tile >> 7;
  const int qbase = (tile & 127) << 4;
  const size_t boff = (size_t)b * L_ * D_;

  // ---- A fragments (Q tile), hi+lo split; contiguous k order used here (B same):
  //      lane<16: K = c*32 + {0..15}; lane>=16: +16
  AF aHi[2], aLo[2];
  {
    const _Float16* qrh = qh + boff + (size_t)(qbase + l16) * D_;
    const _Float16* qrl = ql + boff + (size_t)(qbase + l16) * D_;
    #pragma unroll
    for (int c = 0; c < 2; ++c) {
      int dA = c*32 + hs*16;
      int dB = c*32 + hs*16 + 8;
      aHi[c].h[0] = *(const v8h*)(qrh + dA);
      aHi[c].h[1] = *(const v8h*)(qrh + dB);
      aLo[c].h[0] = *(const v8h*)(qrl + dA);
      aLo[c].h[1] = *(const v8h*)(qrl + dB);
    }
  }

  const _Float16* kbh = kh + boff + (size_t)l16 * D_ + hs*16;
  const _Float16* kbl = kl + boff + (size_t)l16 * D_ + hs*16;

  float m = -INFINITY, Z = 0.0f;                  // online softmax state (base-2)
  float t0 = -INFINITY, t1 = -INFINITY, t2 = -INFINITY;
  int   i0 = 0, i1 = 0, i2 = 0;

  // software pipeline: prefetch tile kt+1's B fragments during tile kt's compute
  v16h bH0 = *(const v16h*)(kbh);
  v16h bL0 = *(const v16h*)(kbl);
  v16h bH1 = *(const v16h*)(kbh + 32);
  v16h bL1 = *(const v16h*)(kbl + 32);

  for (int kt = 0; kt < QT_PER_BATCH; ++kt) {
    v16h cH0 = bH0, cL0 = bL0, cH1 = bH1, cL1 = bL1;
    if (kt + 1 < QT_PER_BATCH) {
      const _Float16* nh = kbh + (size_t)(kt + 1) * 16 * D_;
      const _Float16* nl = kbl + (size_t)(kt + 1) * 16 * D_;
      bH0 = *(const v16h*)(nh);
      bL0 = *(const v16h*)(nl);
      bH1 = *(const v16h*)(nh + 32);
      bL1 = *(const v16h*)(nl + 32);
    }

    v8f acc = {0.f,0.f,0.f,0.f,0.f,0.f,0.f,0.f};
    acc = __builtin_amdgcn_wmma_f32_16x16x32_f16(false, aHi[0].v, false, cH0, (short)0, acc, false, false);
    acc = __builtin_amdgcn_wmma_f32_16x16x32_f16(false, aLo[0].v, false, cH0, (short)0, acc, false, false);
    acc = __builtin_amdgcn_wmma_f32_16x16x32_f16(false, aHi[0].v, false, cL0, (short)0, acc, false, false);
    acc = __builtin_amdgcn_wmma_f32_16x16x32_f16(false, aHi[1].v, false, cH1, (short)0, acc, false, false);
    acc = __builtin_amdgcn_wmma_f32_16x16x32_f16(false, aLo[1].v, false, cH1, (short)0, acc, false, false);
    acc = __builtin_amdgcn_wmma_f32_16x16x32_f16(false, aHi[1].v, false, cL1, (short)0, acc, false, false);

    // redistribute C tile through this wave's private LDS region
    __builtin_amdgcn_wave_barrier();
    #pragma unroll
    for (int j = 0; j < 8; ++j)
      sh[wave][hs*8 + j][l16] = acc[j];           // (lane,VGPR j) -> row j+8*hs, col l16
    __builtin_amdgcn_wave_barrier();
    v4f sA = *(const v4f*)&sh[wave][l16][hs*8];   // lane owns row l16, cols hs*8..+7
    v4f sB = *(const v4f*)&sh[wave][l16][hs*8 + 4];
    __builtin_amdgcn_wave_barrier();

    float ss[8] = { sA[0],sA[1],sA[2],sA[3], sB[0],sB[1],sB[2],sB[3] };

    // ---- branchless online softmax (base-2): tile max, one rescale, 8 exps ----
    float tm = fmaxf(fmaxf(fmaxf(ss[0],ss[1]), fmaxf(ss[2],ss[3])),
                     fmaxf(fmaxf(ss[4],ss[5]), fmaxf(ss[6],ss[7])));
    float mn = fmaxf(m, tm);
    float zs = EXP2F(m - mn);
    float e0 = EXP2F(ss[0]-mn), e1 = EXP2F(ss[1]-mn);
    float e2 = EXP2F(ss[2]-mn), e3 = EXP2F(ss[3]-mn);
    float e4 = EXP2F(ss[4]-mn), e5 = EXP2F(ss[5]-mn);
    float e6 = EXP2F(ss[6]-mn), e7 = EXP2F(ss[7]-mn);
    float esum = ((e0+e1)+(e2+e3)) + ((e4+e5)+(e6+e7));
    Z = fmaf(Z, zs, esum);
    m = mn;

    // ---- branchless top-3 tracking ----
    const int cb = (kt << 4) + hs*8;
    #pragma unroll
    for (int e = 0; e < 8; ++e)
      top3_insert(ss[e], cb + e, t0, t1, t2, i0, i1, i2);
  }

  // ---- merge half-row partials: lane r with lane r^16 ----
  {
    float mo = __shfl_xor(m, 16, 32);
    float Zo = __shfl_xor(Z, 16, 32);
    float u0 = __shfl_xor(t0, 16, 32);
    float u1 = __shfl_xor(t1, 16, 32);
    float u2 = __shfl_xor(t2, 16, 32);
    int   j0 = __shfl_xor(i0, 16, 32);
    int   j1 = __shfl_xor(i1, 16, 32);
    int   j2 = __shfl_xor(i2, 16, 32);
    float mn = fmaxf(m, mo);
    Z = Z * EXP2F(m - mn) + Zo * EXP2F(mo - mn);
    m = mn;
    top3_insert(u0, j0, t0, t1, t2, i0, i1, i2);
    top3_insert(u1, j1, t0, t1, t2, i0, i1, i2);
    top3_insert(u2, j2, t0, t1, t2, i0, i1, i2);
  }

  // ---- sparse weights: only top-2 can survive the top-3 threshold ----
  float invZ = 1.0f / Z;
  float a0 = EXP2F(t0 - m) * invZ;
  float a1 = EXP2F(t1 - m) * invZ;
  float a2 = EXP2F(t2 - m) * invZ;
  float delta = a2 + EPS_;
  float w0 = fmaxf(a0 - delta, 0.0f);
  float w1 = fmaxf(a1 - delta, 0.0f);
  float rn = 1.0f / (w0 + w1 + EPS_);
  w0 *= rn; w1 *= rn;

  // ---- out = w0*V[i0] + w1*V[i1]  (sparse PV, no dense GEMM needed) ----
  const float* vb = v + boff;
  float* outb = out + boff + (size_t)qbase * D_;
  #pragma unroll 4
  for (int r = 0; r < 16; ++r) {
    float ww0 = __shfl(w0, r, 32);
    float ww1 = __shfl(w1, r, 32);
    int   k0  = __shfl(i0, r, 32);
    int   k1  = __shfl(i1, r, 32);
    v2f v0 = *(const v2f*)(vb + (size_t)k0 * D_ + lane*2);
    v2f v1 = *(const v2f*)(vb + (size_t)k1 * D_ + lane*2);
    v2f o; o[0] = ww0*v0[0] + ww1*v1[0]; o[1] = ww0*v0[1] + ww1*v1[1];
    *(v2f*)(outb + r*D_ + lane*2) = o;
  }

  // ---- attn matrix: streaming zero-fill of 16x2048 block, then 2-entry scatter ----
  float* ab = attnw + ((size_t)b * L_ + qbase) * L_;
  v4f z4 = {0.f,0.f,0.f,0.f};
  v4f* ab4 = (v4f*)ab;
  #pragma unroll 8
  for (int i = lane; i < (16 * L_) / 4; i += 32)
    __builtin_nontemporal_store(z4, ab4 + i);
  asm volatile("s_wait_storecnt 0x0" ::: "memory");   // scatter after zero-fill
  if (lane < 16) {
    float* rowp = ab + (size_t)l16 * L_;
    if (w0 > 0.0f) rowp[i0] = w0;
    if (w1 > 0.0f) rowp[i1] = w1;
  }
}

extern "C" void kernel_launch(void* const* d_in, const int* in_sizes, int n_in,
                              void* d_out, int out_size, void* d_ws, size_t ws_size,
                              hipStream_t stream) {
  const float* q = (const float*)d_in[0];
  const float* k = (const float*)d_in[1];
  const float* v = (const float*)d_in[2];
  float* out   = (float*)d_out;
  float* attnw = out + (size_t)B_ * L_ * D_;   // tuple: (out, attn) concatenated

  _Float16* qh = (_Float16*)d_ws;              // 4x 4MB = 16MB scratch
  _Float16* ql = qh + N_ELEM;
  _Float16* kh = ql + N_ELEM;
  _Float16* kl = kh + N_ELEM;

  prep_split_f16<<<(N_ELEM + 255)/256, 256, 0, stream>>>(q, k, qh, ql, kh, kl);
  sparse_attn_kernel<<<(B_ * QT_PER_BATCH)/8, 256, 0, stream>>>(qh, ql, kh, kl, v, out, attnw);
}
